// TrmEncoder_17712445129039
// MI455X (gfx1250) — compile-verified, matrix operand order checked by reading, not run on hardware
//
#include <hip/hip_runtime.h>

#define NSEQ   3072
#define DMODEL 256
#define NHEAD  8
#define HDK    256
#define HD     (NHEAD * HDK)   // 2048
#define NEGF   (-4294967295.0f)
#define LOG2E  1.4426950408889634f

typedef __attribute__((ext_vector_type(16))) __bf16 v16bf;
typedef __attribute__((ext_vector_type(8)))  float  v8f;

union FragBF { v16bf v; unsigned int u[8]; };

__device__ __forceinline__ unsigned short f2bf(float f) {
  unsigned int u = __float_as_uint(f);
  u += 0x7fffu + ((u >> 16) & 1u);     // round to nearest even
  return (unsigned short)(u >> 16);
}

__device__ __forceinline__ v8f v8zero() {
  v8f z;
#pragma unroll
  for (int i = 0; i < 8; ++i) z[i] = 0.0f;
  return z;
}

// ---- CDNA5 async global->LDS copy (ASYNCcnt-tracked, no VGPR data path) ----
// dst: generic pointer into __shared__ (low 32 bits = LDS byte offset)
// src: 16B-aligned global address; moves 16 bytes per active lane.
__device__ __forceinline__ void async_ld_b128(void* lds_dst, const void* gsrc) {
  unsigned int loff = (unsigned int)(unsigned long long)lds_dst;
  asm volatile("global_load_async_to_lds_b128 %0, %1, off"
               :: "v"(loff), "v"(gsrc) : "memory");
}
__device__ __forceinline__ void wait_async0() {
  asm volatile("s_wait_asynccnt 0" ::: "memory");
}

// Gather a 16x32 bf16 WMMA fragment from LDS. For lane l: row/col = l%16,
// K-halves at kb=(l/16)*8 : elements {kb..kb+7, kb+16..kb+23}, pairs dword-contiguous.
__device__ __forceinline__ void load_frag(FragBF& f, const unsigned short* p) {
#pragma unroll
  for (int i = 0; i < 4; ++i) f.u[i]     = *(const unsigned int*)(p + 2 * i);
#pragma unroll
  for (int i = 0; i < 4; ++i) f.u[4 + i] = *(const unsigned int*)(p + 16 + 2 * i);
}

// ---------------------------------------------------------------------------
// features[n][d] = port_embed[up[n]][d] + weights[n]*we_w[d] + we_b[d]
// ---------------------------------------------------------------------------
__global__ __launch_bounds__(256) void feat_kernel(
    const float* __restrict__ weights, const int* __restrict__ up,
    const float* __restrict__ pe, const float* __restrict__ we_w,
    const float* __restrict__ we_b, float* __restrict__ featF,
    unsigned short* __restrict__ featB)
{
  const int n = blockIdx.x, d = threadIdx.x;
  float v = pe[up[n] * DMODEL + d] + weights[n] * we_w[d] + we_b[d];
  featF[(size_t)n * DMODEL + d] = v;
  featB[(size_t)n * DMODEL + d] = f2bf(v);
}

// fp32 -> bf16 convert, optional transpose (out[c][r] = in[r][c])
__global__ __launch_bounds__(256) void conv_bf16(
    const float* __restrict__ in, unsigned short* __restrict__ out,
    int R, int C, int transpose)
{
  int idx = blockIdx.x * 256 + threadIdx.x;
  if (idx >= R * C) return;
  int r = idx / C, c = idx % C;
  unsigned short b = f2bf(in[idx]);
  if (transpose) out[(size_t)c * R + r] = b;
  else           out[idx] = b;
}

// ---------------------------------------------------------------------------
// bf16 GEMM: C[M,Nc] = A[M,K] (row-major bf16) @ B[K,Nc] (row-major bf16)
// block tile 128x64, 8 waves (4x2) of 2x2 WMMA 16x16 tiles, K-step 32.
// MODE 0: bf16 out; 1: f32 out; 2: bf16 out with ReLU. HAS_BIAS adds biasVec.
// A tile staged via global_load_async_to_lds_b128.
// ---------------------------------------------------------------------------
#define GTM 128
#define GTN 64
#define GTK 32
#define GLA 40    // 80B rows: 16B-aligned for async B128, bank-conflict-free
#define GLB 34

template <int MODE, bool HAS_BIAS>
__global__ __launch_bounds__(256) void gemm_bf16_kernel(
    const unsigned short* __restrict__ A, const unsigned short* __restrict__ B,
    void* __restrict__ Cout, const float* __restrict__ biasVec,
    int M, int Nc, int Kd)
{
  (void)M;
  __shared__ __align__(16) unsigned short sA[GTM * GLA];
  __shared__ __align__(16) unsigned short sB[GTN * GLB];   // transposed: [n][k]
  const int tid = threadIdx.x;
  const int lane = tid & 31;
  const int wave = tid >> 5;
  const int wm = wave & 3, wn = wave >> 2;
  const int m0 = blockIdx.x * GTM;
  const int n0 = blockIdx.y * GTN;
  const int lm = lane & 15, hg = lane >> 4, kb = hg * 8;

  v8f acc[2][2];
#pragma unroll
  for (int i = 0; i < 2; ++i)
#pragma unroll
    for (int j = 0; j < 2; ++j) acc[i][j] = v8zero();

  for (int kt = 0; kt < Kd; kt += GTK) {
    // A tile 128x32 halfs: 512 async b128 transfers, 2 per thread
#pragma unroll
    for (int i = 0; i < 2; ++i) {
      int v = tid + i * 256;
      int row = v >> 2;
      int o8  = (v & 3) * 8;
      async_ld_b128(sA + row * GLA + o8,
                    A + (size_t)(m0 + row) * Kd + kt + o8);
    }
    // B tile 32x64 halfs, store transposed so fragment K-pairs are contiguous
    {
      int krow = tid >> 3;
      int noff = (tid & 7) * 8;
      uint4 d = *(const uint4*)(B + (size_t)(kt + krow) * Nc + n0 + noff);
      unsigned int w[4] = { d.x, d.y, d.z, d.w };
#pragma unroll
      for (int j = 0; j < 4; ++j) {
        sB[(noff + 2 * j    ) * GLB + krow] = (unsigned short)(w[j] & 0xffffu);
        sB[(noff + 2 * j + 1) * GLB + krow] = (unsigned short)(w[j] >> 16);
      }
    }
    wait_async0();
    __syncthreads();

    FragBF af[2], bf[2];
#pragma unroll
    for (int mi = 0; mi < 2; ++mi)
      load_frag(af[mi], sA + (wm * 32 + mi * 16 + lm) * GLA + kb);
#pragma unroll
    for (int ni = 0; ni < 2; ++ni)
      load_frag(bf[ni], sB + (wn * 32 + ni * 16 + lm) * GLB + kb);
#pragma unroll
    for (int mi = 0; mi < 2; ++mi)
#pragma unroll
      for (int ni = 0; ni < 2; ++ni)
        acc[mi][ni] = __builtin_amdgcn_wmma_f32_16x16x32_bf16(
            false, af[mi].v, false, bf[ni].v, (short)0, acc[mi][ni], false, false);
    __syncthreads();
  }

#pragma unroll
  for (int mi = 0; mi < 2; ++mi)
#pragma unroll
    for (int ni = 0; ni < 2; ++ni) {
      int col = n0 + wn * 32 + ni * 16 + lm;
      float bv = 0.0f;
      if constexpr (HAS_BIAS) bv = biasVec[col];
#pragma unroll
      for (int r = 0; r < 8; ++r) {
        int row = m0 + wm * 32 + mi * 16 + 8 * hg + r;
        float v = acc[mi][ni][r] + bv;
        if constexpr (MODE == 2) v = fmaxf(v, 0.0f);
        if constexpr (MODE == 1)
          ((float*)Cout)[(size_t)row * Nc + col] = v;
        else
          ((unsigned short*)Cout)[(size_t)row * Nc + col] = f2bf(v);
      }
    }
}

// ---------------------------------------------------------------------------
// Flash attention with DUAL online softmax (scores + bias), causal.
// Block = 32 q-rows x 1 head, 8 waves: wq = wave&1 (16-row subtile),
// wv = wave>>1 (64-wide dv quarter). Waves with wv==0 own the softmax.
// Q/K tiles staged via async b128; V transposed through registers.
// ---------------------------------------------------------------------------
#define AQ  32
#define AKB 32
#define QLP 264   // 528B rows: 16B-aligned for async B128, conflict-free
#define KLP 264
#define VLP 34
#define PLP 34

__device__ __forceinline__ void online_softmax(
    v8f (&s)[2], float (&m)[8], float (&l)[8], float (&alpha)[8],
    unsigned short* __restrict__ pdst, int lm, int hg)
{
  float rmax[8];
#pragma unroll
  for (int r = 0; r < 8; ++r) rmax[r] = fmaxf(s[0][r], s[1][r]);
#pragma unroll
  for (int o = 1; o < 16; o <<= 1)
#pragma unroll
    for (int r = 0; r < 8; ++r) rmax[r] = fmaxf(rmax[r], __shfl_xor(rmax[r], o, 32));
#pragma unroll
  for (int r = 0; r < 8; ++r) {
    float mn = fmaxf(m[r], rmax[r]);
    alpha[r] = exp2f((m[r] - mn) * LOG2E);
    m[r] = mn;
  }
#pragma unroll
  for (int t = 0; t < 2; ++t)
#pragma unroll
    for (int r = 0; r < 8; ++r) s[t][r] = exp2f((s[t][r] - m[r]) * LOG2E);
  float rsum[8];
#pragma unroll
  for (int r = 0; r < 8; ++r) rsum[r] = s[0][r] + s[1][r];
#pragma unroll
  for (int o = 1; o < 16; o <<= 1)
#pragma unroll
    for (int r = 0; r < 8; ++r) rsum[r] += __shfl_xor(rsum[r], o, 32);
#pragma unroll
  for (int r = 0; r < 8; ++r) l[r] = l[r] * alpha[r] + rsum[r];
  // Stage P (accumulator layout -> LDS) for re-read in A-fragment layout
#pragma unroll
  for (int t = 0; t < 2; ++t)
#pragma unroll
    for (int r = 0; r < 8; ++r)
      pdst[(8 * hg + r) * PLP + t * 16 + lm] = f2bf(s[t][r]);
}

__global__ __launch_bounds__(256) void attn_kernel(
    const unsigned short* __restrict__ Qg, const unsigned short* __restrict__ Kg,
    const unsigned short* __restrict__ Vg, const float* __restrict__ bias,
    unsigned short* __restrict__ AO)
{
  __shared__ __align__(16) unsigned short sQ[AQ * QLP];
  __shared__ __align__(16) unsigned short sK[AKB * KLP];
  __shared__ __align__(16) unsigned short sVt[256 * VLP];      // [dv][key]
  __shared__ __align__(16) unsigned short sP[2][2][16 * PLP];  // [wq][softmax]
  __shared__ float sAlpha[2][2][16];
  __shared__ float sL[2][2][16];

  const int tid = threadIdx.x;
  const int lane = tid & 31;
  const int wave = tid >> 5;
  const int wq = wave & 1;
  const int wv = wave >> 1;
  const int h  = blockIdx.y;
  const int q0 = blockIdx.x * AQ;
  const int lm = lane & 15, hg = lane >> 4, kb8 = hg * 8;
  const float scale = 1.0f / 16.000001f;   // 1/(sqrt(256)+1e-6)

  // Load Q tile (AQ x 256 bf16) once, asynchronously
#pragma unroll
  for (int i = 0; i < 4; ++i) {
    int v = tid + i * 256;
    int row = v >> 5;
    int o8  = (v & 31) * 8;
    async_ld_b128(sQ + row * QLP + o8,
                  Qg + (size_t)(q0 + row) * HD + h * HDK + o8);
  }

  v8f o1[4], o2[4];
#pragma unroll
  for (int nt = 0; nt < 4; ++nt) { o1[nt] = v8zero(); o2[nt] = v8zero(); }
  float m1[8], l1[8], m2[8], l2[8];
#pragma unroll
  for (int r = 0; r < 8; ++r) { m1[r] = -3.0e38f; m2[r] = -3.0e38f; l1[r] = 0.f; l2[r] = 0.f; }

  const int nkb = (q0 + AQ - 1) / AKB + 1;   // causal: keys up to q0+AQ-1
  for (int kbi = 0; kbi < nkb; ++kbi) {
    const int k0 = kbi * AKB;
    __syncthreads();
    // K tile (AKB x 256) row-major via async copy
#pragma unroll
    for (int i = 0; i < 4; ++i) {
      int v = tid + i * 256;
      int row = v >> 5;
      int o8  = (v & 31) * 8;
      async_ld_b128(sK + row * KLP + o8,
                    Kg + (size_t)(k0 + row) * HD + h * HDK + o8);
    }
    // V tile transposed into [dv][key] (register-mediated transpose)
#pragma unroll
    for (int i = 0; i < 4; ++i) {
      int v = tid + i * 256;
      int row = v >> 5;
      int o8  = (v & 31) * 8;
      uint4 d = *(const uint4*)(Vg + (size_t)(k0 + row) * HD + h * HDK + o8);
      unsigned int w[4] = { d.x, d.y, d.z, d.w };
#pragma unroll
      for (int j = 0; j < 4; ++j) {
        sVt[(o8 + 2 * j    ) * VLP + row] = (unsigned short)(w[j] & 0xffffu);
        sVt[(o8 + 2 * j + 1) * VLP + row] = (unsigned short)(w[j] >> 16);
      }
    }
    wait_async0();
    __syncthreads();

    if (wv == 0) {
      // S = Q @ K^T for this wq subtile (16 x 32 keys), DK=256 -> 8 WMMA k-steps
      v8f s[2]; s[0] = v8zero(); s[1] = v8zero();
      FragBF qa, kf;
#pragma unroll
      for (int kt = 0; kt < 8; ++kt) {
        load_frag(qa, sQ + (wq * 16 + lm) * QLP + kt * 32 + kb8);
#pragma unroll
        for (int t = 0; t < 2; ++t) {
          load_frag(kf, sK + (t * 16 + lm) * KLP + kt * 32 + kb8);
          s[t] = __builtin_amdgcn_wmma_f32_16x16x32_bf16(
              false, qa.v, false, kf.v, (short)0, s[t], false, false);
        }
      }
      // scale + causal mask; load bias tile (fp32 stream from HBM)
      v8f sb[2];
#pragma unroll
      for (int t = 0; t < 2; ++t) {
        int key = k0 + t * 16 + lm;
#pragma unroll
        for (int r = 0; r < 8; ++r) {
          int qr = q0 + wq * 16 + 8 * hg + r;
          float sv = s[t][r] * scale;
          float bv = bias[(size_t)qr * NSEQ + key];
          if (key > qr) { sv = NEGF; bv = NEGF; }
          s[t][r] = sv;
          sb[t][r] = bv;
        }
      }
      float a1[8], a2[8];
      online_softmax(s,  m1, l1, a1, &sP[wq][0][0], lm, hg);
      online_softmax(sb, m2, l2, a2, &sP[wq][1][0], lm, hg);
      if (lm == 0) {
#pragma unroll
        for (int r = 0; r < 8; ++r) {
          sAlpha[wq][0][8 * hg + r] = a1[r];
          sAlpha[wq][1][8 * hg + r] = a2[r];
        }
      }
      if (kbi + 1 < nkb)   // prefetch next bias block -> global_prefetch_b8
        __builtin_prefetch(bias + (size_t)(q0 + wq * 16 + 8 * hg) * NSEQ + k0 + AKB, 0, 0);
    }
    __syncthreads();

    // All waves: rescale O by alpha, accumulate O += P @ V (both softmaxes)
    float a1[8], a2[8];
#pragma unroll
    for (int r = 0; r < 8; ++r) {
      a1[r] = sAlpha[wq][0][8 * hg + r];
      a2[r] = sAlpha[wq][1][8 * hg + r];
    }
#pragma unroll
    for (int nt = 0; nt < 4; ++nt)
#pragma unroll
      for (int r = 0; r < 8; ++r) { o1[nt][r] *= a1[r]; o2[nt][r] *= a2[r]; }

    FragBF p1, p2, vf;
    load_frag(p1, &sP[wq][0][0] + lm * PLP + kb8);
    load_frag(p2, &sP[wq][1][0] + lm * PLP + kb8);
#pragma unroll
    for (int nt = 0; nt < 4; ++nt) {
      load_frag(vf, sVt + (wv * 64 + nt * 16 + lm) * VLP + kb8);
      o1[nt] = __builtin_amdgcn_wmma_f32_16x16x32_bf16(
          false, p1.v, false, vf.v, (short)0, o1[nt], false, false);
      o2[nt] = __builtin_amdgcn_wmma_f32_16x16x32_bf16(
          false, p2.v, false, vf.v, (short)0, o2[nt], false, false);
    }
  }

  __syncthreads();
  if (wv == 0 && lm == 0) {
#pragma unroll
    for (int r = 0; r < 8; ++r) {
      sL[wq][0][8 * hg + r] = l1[r];
      sL[wq][1][8 * hg + r] = l2[r];
    }
  }
  __syncthreads();
#pragma unroll
  for (int nt = 0; nt < 4; ++nt)
#pragma unroll
    for (int r = 0; r < 8; ++r) {
      int qr = q0 + wq * 16 + 8 * hg + r;
      int dv = wv * 64 + nt * 16 + lm;
      float rl1 = sL[wq][0][8 * hg + r];
      float rl2 = sL[wq][1][8 * hg + r];
      float val = o1[nt][r] / rl1 + o2[nt][r] / rl2;
      AO[(size_t)qr * HD + h * HDK + dv] = f2bf(val);
    }
}

// ---------------------------------------------------------------------------
// LayerNorm over rows of 256: out = LN(a + b) * g + beta  (b optional)
// ---------------------------------------------------------------------------
__global__ __launch_bounds__(256) void ln_kernel(
    const float* __restrict__ a, const float* __restrict__ b,
    const float* __restrict__ g, const float* __restrict__ beta,
    float* __restrict__ outF, unsigned short* __restrict__ outB)
{
  __shared__ float sh[256];
  __shared__ float stat;
  const int row = blockIdx.x, d = threadIdx.x;
  float x = a[(size_t)row * DMODEL + d] + (b ? b[(size_t)row * DMODEL + d] : 0.0f);
  sh[d] = x; __syncthreads();
  for (int s = 128; s > 0; s >>= 1) { if (d < s) sh[d] += sh[d + s]; __syncthreads(); }
  if (d == 0) stat = sh[0] * (1.0f / DMODEL);
  __syncthreads();
  float mu = stat;
  float c = x - mu;
  sh[d] = c * c; __syncthreads();
  for (int s = 128; s > 0; s >>= 1) { if (d < s) sh[d] += sh[d + s]; __syncthreads(); }
  if (d == 0) stat = sh[0] * (1.0f / DMODEL);
  __syncthreads();
  float y = c * rsqrtf(stat + 1e-5f) * g[d] + beta[d];
  outF[(size_t)row * DMODEL + d] = y;
  if (outB) outB[(size_t)row * DMODEL + d] = f2bf(y);
}

// ---------------------------------------------------------------------------
extern "C" void kernel_launch(void* const* d_in, const int* in_sizes, int n_in,
                              void* d_out, int out_size, void* d_ws, size_t ws_size,
                              hipStream_t stream)
{
  (void)in_sizes; (void)n_in; (void)out_size; (void)ws_size;
  const float* weights = (const float*)d_in[0];
  const int*   up      = (const int*)d_in[1];
  const float* bias    = (const float*)d_in[2];
  const float* pe      = (const float*)d_in[3];
  const float* we_w    = (const float*)d_in[4];
  const float* we_b    = (const float*)d_in[5];
  const float* W_q     = (const float*)d_in[6];
  const float* W_k     = (const float*)d_in[7];
  const float* W_v     = (const float*)d_in[8];
  const float* W_o     = (const float*)d_in[9];
  const float* l1_w    = (const float*)d_in[10];
  const float* l1_b    = (const float*)d_in[11];
  const float* l2_w    = (const float*)d_in[12];
  const float* l2_b    = (const float*)d_in[13];
  const float* ln_g    = (const float*)d_in[14];
  const float* ln_b    = (const float*)d_in[15];

  char* ws = (char*)d_ws;
  size_t off = 0;
  auto alloc = [&](size_t bytes) -> void* {
    void* p = ws + off;
    off += (bytes + 255) & ~(size_t)255;
    return p;
  };

  float*          featF = (float*)alloc((size_t)NSEQ * DMODEL * 4);
  unsigned short* featB = (unsigned short*)alloc((size_t)NSEQ * DMODEL * 2);
  unsigned short* WqB   = (unsigned short*)alloc((size_t)DMODEL * HD * 2);
  unsigned short* WkB   = (unsigned short*)alloc((size_t)DMODEL * HD * 2);
  unsigned short* WvB   = (unsigned short*)alloc((size_t)DMODEL * HD * 2);
  unsigned short* WoB   = (unsigned short*)alloc((size_t)HD * DMODEL * 2);
  unsigned short* l1B   = (unsigned short*)alloc((size_t)DMODEL * DMODEL * 2);
  unsigned short* l2B   = (unsigned short*)alloc((size_t)DMODEL * DMODEL * 2);
  unsigned short* Qb    = (unsigned short*)alloc((size_t)NSEQ * HD * 2);
  unsigned short* Kb    = (unsigned short*)alloc((size_t)NSEQ * HD * 2);
  unsigned short* Vb    = (unsigned short*)alloc((size_t)NSEQ * HD * 2);
  unsigned short* AO    = (unsigned short*)alloc((size_t)NSEQ * HD * 2);
  float*          T1    = (float*)alloc((size_t)NSEQ * DMODEL * 4);
  float*          Xf    = (float*)alloc((size_t)NSEQ * DMODEL * 4);
  unsigned short* Xb    = (unsigned short*)alloc((size_t)NSEQ * DMODEL * 2);
  unsigned short* F1    = (unsigned short*)alloc((size_t)NSEQ * DMODEL * 2);
  float*          T2    = (float*)alloc((size_t)NSEQ * DMODEL * 4);

  // 1. features + weight conversion
  feat_kernel<<<NSEQ, 256, 0, stream>>>(weights, up, pe, we_w, we_b, featF, featB);
  conv_bf16<<<(DMODEL * HD) / 256, 256, 0, stream>>>(W_q, WqB, DMODEL, HD, 0);
  conv_bf16<<<(DMODEL * HD) / 256, 256, 0, stream>>>(W_k, WkB, DMODEL, HD, 0);
  conv_bf16<<<(DMODEL * HD) / 256, 256, 0, stream>>>(W_v, WvB, DMODEL, HD, 0);
  conv_bf16<<<(HD * DMODEL) / 256, 256, 0, stream>>>(W_o, WoB, HD, DMODEL, 0);
  conv_bf16<<<(DMODEL * DMODEL) / 256, 256, 0, stream>>>(l1_w, l1B, DMODEL, DMODEL, 1);
  conv_bf16<<<(DMODEL * DMODEL) / 256, 256, 0, stream>>>(l2_w, l2B, DMODEL, DMODEL, 1);

  // 2. QKV projections (WMMA, async-LDS staging)
  dim3 gQKV(NSEQ / GTM, HD / GTN);
  gemm_bf16_kernel<0, false><<<gQKV, 256, 0, stream>>>(featB, WqB, Qb, nullptr, NSEQ, HD, DMODEL);
  gemm_bf16_kernel<0, false><<<gQKV, 256, 0, stream>>>(featB, WkB, Kb, nullptr, NSEQ, HD, DMODEL);
  gemm_bf16_kernel<0, false><<<gQKV, 256, 0, stream>>>(featB, WvB, Vb, nullptr, NSEQ, HD, DMODEL);

  // 3. dual-softmax causal flash attention (WMMA)
  dim3 gA(NSEQ / AQ, NHEAD);
  attn_kernel<<<gA, 256, 0, stream>>>(Qb, Kb, Vb, bias, AO);

  // 4. output projection, residual + LN
  dim3 gO(NSEQ / GTM, DMODEL / GTN);
  gemm_bf16_kernel<1, false><<<gO, 256, 0, stream>>>(AO, WoB, T1, nullptr, NSEQ, DMODEL, HD);
  ln_kernel<<<NSEQ, 256, 0, stream>>>(T1, featF, ln_g, ln_b, Xf, Xb);

  // 5. FFN (WMMA) + residual + LN -> output (fp32)
  gemm_bf16_kernel<2, true><<<gO, 256, 0, stream>>>(Xb, l1B, F1, l1_b, NSEQ, DMODEL, DMODEL);
  gemm_bf16_kernel<1, true><<<gO, 256, 0, stream>>>(F1, l2B, T2, l2_b, NSEQ, DMODEL, DMODEL);
  ln_kernel<<<NSEQ, 256, 0, stream>>>(T2, Xf, ln_g, ln_b, (float*)d_out, nullptr);
}
